// MultiHeadAttention_35459249996373
// MI455X (gfx1250) — matrix operand order repaired and checked by reading, not compiled, not run on hardware
//
#include <hip/hip_runtime.h>

typedef __attribute__((ext_vector_type(16))) _Float16 v16h;
typedef __attribute__((ext_vector_type(8)))  _Float16 v8h;
typedef __attribute__((ext_vector_type(8)))  float    v8f;
typedef __attribute__((ext_vector_type(4)))  int      v4i;

static constexpr int D_MODEL  = 1024;
static constexpr int N_HEADS  = 16;
static constexpr int HEAD_DIM = 64;
static constexpr int BATCH    = 4;
static constexpr int SEQ      = 2048;
static constexpr float SCALE  = 0.125f;   // 1/sqrt(64)

// ---- CDNA5 async global->LDS copy (probe-confirmed signature) ----
#if __has_builtin(__builtin_amdgcn_global_load_async_to_lds_b128)
#define HAVE_ASYNC_LDS 1
typedef __attribute__((address_space(1))) v4i gv4i;
typedef __attribute__((address_space(3))) v4i lv4i;
__device__ __forceinline__ void async_copy16(const void* g, void* l) {
  __builtin_amdgcn_global_load_async_to_lds_b128((gv4i*)g, (lv4i*)l, 0, 0);
}
#else
#define HAVE_ASYNC_LDS 0
#endif

__device__ __forceinline__ void wait_asynccnt0() {
#if __has_builtin(__builtin_amdgcn_s_wait_asynccnt)
  __builtin_amdgcn_s_wait_asynccnt(0);
#else
  asm volatile("s_wait_asynccnt 0" ::: "memory");
#endif
}

__device__ __forceinline__ v8f wmma_f16(v16h a, v16h b, v8f c) {
  return __builtin_amdgcn_wmma_f32_16x16x32_f16(false, a, false, b, (short)0, c, false, false);
}

__device__ __forceinline__ v16h combine(v8h a, v8h b) {
  v16h r;
#pragma unroll
  for (int i = 0; i < 8; ++i) { r[i] = a[i]; r[i + 8] = b[i]; }
  return r;
}

// ---------------- f32 -> f16 cast (activations) ----------------
__global__ void cast_f16_kernel(const float* __restrict__ in, _Float16* __restrict__ out, int n) {
  int i = blockIdx.x * blockDim.x + threadIdx.x;
  if (i < n) out[i] = (_Float16)in[i];
}

// ---------------- f32 -> f16 cast + transpose (weights, one-time) ----------------
// in: [R][C] f32 row-major; out: [C][R] f16 row-major (out[c][r] = in[r][c])
__global__ __launch_bounds__(256) void cast_transpose_kernel(const float* __restrict__ in,
                                                             _Float16* __restrict__ out,
                                                             int R, int C) {
  __shared__ _Float16 tile[32][33];
  const int c0 = blockIdx.x * 32, r0 = blockIdx.y * 32;
  const int lx = threadIdx.x, ly = threadIdx.y;   // block (32, 8)
#pragma unroll
  for (int i = 0; i < 32; i += 8)
    tile[ly + i][lx] = (_Float16)in[(size_t)(r0 + ly + i) * C + c0 + lx];
  __syncthreads();
#pragma unroll
  for (int i = 0; i < 32; i += 8)
    out[(size_t)(c0 + ly + i) * R + r0 + lx] = tile[lx][ly + i];
}

// ---------------- WMMA GEMM: C[M,N] = A[M,K] * Bt[N,K]^T, f16 in, f32 accum ----------------
// Block 256 threads (8 waves), tile 128x128; wave tile 32x64; K chunks of 32.
// Double-buffered LDS stages; both tiles are pure async global->LDS b128 copies.
template <bool OUT_F32>
__global__ __launch_bounds__(256) void gemm_wmma_kernel(const _Float16* __restrict__ A,
                                                        const _Float16* __restrict__ Bt,
                                                        float* __restrict__ Cf,
                                                        _Float16* __restrict__ Ch,
                                                        int M, int N, int K) {
  __shared__ __align__(16) _Float16 sA [2][128 * 40];   // [m][k], stride 40
  __shared__ __align__(16) _Float16 sBt[2][128 * 40];   // [n][k], stride 40
  const int t    = threadIdx.x;
  const int lane = t & 31;
  const int wid  = t >> 5;
  const int hi   = lane >> 4;
  const int cn   = lane & 15;
  const int m0   = blockIdx.y * 128;
  const int n0   = blockIdx.x * 128;
  const int wm   = (wid & 3) * 32;
  const int wn   = (wid >> 2) * 64;

  v8f acc[2][4] = {};

  const int arow = t >> 2;          // 64 rows per pass (2 passes)
  const int acol = (t & 3) * 8;

  auto load_tiles = [&](int kc, int st) {
#pragma unroll
    for (int q = 0; q < 2; ++q) {
      int row = q * 64 + arow;
#if HAVE_ASYNC_LDS
      async_copy16(A  + (size_t)(m0 + row) * K + kc + acol, &sA [st][row * 40 + acol]);
      async_copy16(Bt + (size_t)(n0 + row) * K + kc + acol, &sBt[st][row * 40 + acol]);
#else
      *(uint4*)&sA [st][row * 40 + acol] = *(const uint4*)(A  + (size_t)(m0 + row) * K + kc + acol);
      *(uint4*)&sBt[st][row * 40 + acol] = *(const uint4*)(Bt + (size_t)(n0 + row) * K + kc + acol);
#endif
    }
  };

  load_tiles(0, 0);
  for (int k0 = 0; k0 < K; k0 += 32) {
    const int st = (k0 >> 5) & 1;
    wait_asynccnt0();       // our copies into stage st have landed
    __syncthreads();        // everyone's copies landed; everyone done reading st^1
    if (k0 + 32 < K) load_tiles(k0 + 32, st ^ 1);   // overlaps with compute below

    v16h af[2], bf[4];
#pragma unroll
    for (int mi = 0; mi < 2; ++mi) {
      const _Float16* p = &sA[st][(wm + mi * 16 + cn) * 40] + hi * 8;
      af[mi] = combine(*(const v8h*)p, *(const v8h*)(p + 16));
    }
#pragma unroll
    for (int j = 0; j < 4; ++j) {
      const _Float16* p = &sBt[st][(wn + j * 16 + cn) * 40] + hi * 8;
      bf[j] = combine(*(const v8h*)p, *(const v8h*)(p + 16));
    }
#pragma unroll
    for (int mi = 0; mi < 2; ++mi)
#pragma unroll
      for (int j = 0; j < 4; ++j)
        acc[mi][j] = wmma_f16(af[mi], bf[j], acc[mi][j]);
  }

#pragma unroll
  for (int mi = 0; mi < 2; ++mi)
#pragma unroll
    for (int j = 0; j < 4; ++j)
#pragma unroll
      for (int r = 0; r < 8; ++r) {
        int row = m0 + wm + mi * 16 + r + hi * 8;
        int col = n0 + wn + j * 16 + cn;
        size_t off = (size_t)row * N + col;
        if (OUT_F32) Cf[off] = acc[mi][j][r];
        else         Ch[off] = (_Float16)acc[mi][j][r];
      }
}

// ---------------- Flash attention (causal + pad), f16 QKV, f32 accum ----------------
// Grid (SEQ/128, BATCH*N_HEADS); 256 threads (8 waves); wave owns 16 query rows.
// Double-buffered K/V stages: K via async copy, V transposed on store.
__global__ __launch_bounds__(256) void flash_attn_kernel(const _Float16* __restrict__ Q,
                                                         const _Float16* __restrict__ Kp,
                                                         const _Float16* __restrict__ Vp,
                                                         const unsigned char* __restrict__ pad,
                                                         _Float16* __restrict__ O) {
  __shared__ __align__(16) _Float16 sK [2][64 * 72];
  __shared__ __align__(16) _Float16 sVt[2][64 * 72];
  __shared__ __align__(16) _Float16 sP [8][16 * 72];
  const int t    = threadIdx.x;
  const int lane = t & 31;
  const int wid  = t >> 5;
  const int hi   = lane >> 4;
  const int cn   = lane & 15;
  const int q0   = blockIdx.x * 128;
  const int b    = blockIdx.y >> 4;
  const int h    = blockIdx.y & 15;
  const size_t rowbase = (size_t)(b * SEQ + q0 + wid * 16);

  // Q tile (16x64) -> 2 A fragments via 4x global b128 per lane
  v16h qf[2];
  {
    const _Float16* qp = Q + (rowbase + cn) * D_MODEL + h * HEAD_DIM;
#pragma unroll
    for (int c = 0; c < 2; ++c) {
      const _Float16* p = qp + c * 32 + hi * 8;
      qf[c] = combine(*(const v8h*)p, *(const v8h*)(p + 16));
    }
  }

  v8f acc[4] = {};
  float mrun[8], lrun[8];
#pragma unroll
  for (int r = 0; r < 8; ++r) { mrun[r] = -1e30f; lrun[r] = 0.0f; }

  const int nkb  = 2 * blockIdx.x + 2;   // causal: up to diagonal block
  const int lrow = t >> 3;               // 32 key rows per pass
  const int lcg  = (t & 7) * 8;

  auto load_kv = [&](int kb, int st) {
#pragma unroll
    for (int p = 0; p < 2; ++p) {
      int row = p * 32 + lrow;
      size_t g = (size_t)(b * SEQ + kb * 64 + row) * D_MODEL + h * HEAD_DIM + lcg;
#if HAVE_ASYNC_LDS
      async_copy16(Kp + g, &sK[st][row * 72 + lcg]);    // K: direct async copy
#else
      *(uint4*)&sK[st][row * 72 + lcg] = *(const uint4*)(Kp + g);
#endif
      union { uint4 u; _Float16 hh[8]; } dv;            // V: transpose on store
      dv.u = *(const uint4*)(Vp + g);
#pragma unroll
      for (int i = 0; i < 8; ++i) sVt[st][(lcg + i) * 72 + row] = dv.hh[i];
    }
  };

  load_kv(0, 0);
  for (int kb = 0; kb < nkb; ++kb) {
    const int st    = kb & 1;
    const int kbase = kb * 64;
    wait_asynccnt0();
    __syncthreads();                    // stage st published; stage st^1 free
    if (kb + 1 < nkb) load_kv(kb + 1, st ^ 1);   // overlaps with compute below

    // ---- S = Q*K^T (8 WMMAs), scale + causal/pad mask ----
    v8f sc[4];
#pragma unroll
    for (int j = 0; j < 4; ++j) {
      v8f s = {};
#pragma unroll
      for (int c = 0; c < 2; ++c) {
        const _Float16* p = &sK[st][(j * 16 + cn) * 72] + c * 32 + hi * 8;
        v16h kf = combine(*(const v8h*)p, *(const v8h*)(p + 16));
        s = wmma_f16(qf[c], kf, s);
      }
      int  kr = kbase + j * 16 + cn;
      bool pm = pad[b * SEQ + kr] != 0;
#pragma unroll
      for (int r = 0; r < 8; ++r) {
        int qr = q0 + wid * 16 + r + hi * 8;
        float v = s[r] * SCALE;
        sc[j][r] = (pm || (kr > qr)) ? -1e30f : v;
      }
    }

    // ---- online softmax: row reductions across 16-lane half ----
#pragma unroll
    for (int r = 0; r < 8; ++r) {
      float tm = fmaxf(fmaxf(sc[0][r], sc[1][r]), fmaxf(sc[2][r], sc[3][r]));
#pragma unroll
      for (int off = 8; off > 0; off >>= 1) tm = fmaxf(tm, __shfl_xor(tm, off, 16));
      float mnew  = fmaxf(mrun[r], tm);
      float alpha = __expf(mrun[r] - mnew);
      mrun[r] = mnew;
      float rs = 0.0f;
#pragma unroll
      for (int j = 0; j < 4; ++j) {
        float p = __expf(sc[j][r] - mnew);
        sP[wid][(r + hi * 8) * 72 + j * 16 + cn] = (_Float16)p;
        rs += p;
      }
#pragma unroll
      for (int off = 8; off > 0; off >>= 1) rs += __shfl_xor(rs, off, 16);
      lrun[r] = lrun[r] * alpha + rs;
#pragma unroll
      for (int j2 = 0; j2 < 4; ++j2) acc[j2][r] *= alpha;
    }

    // wave-local: P stores must land before the A-fragment regather
    asm volatile("s_wait_dscnt 0" ::: "memory");

    // ---- O += P*V (8 WMMAs) ----
#pragma unroll
    for (int c = 0; c < 2; ++c) {
      const _Float16* pp = sP[wid] + cn * 72 + c * 32 + hi * 8;
      v16h pf = combine(*(const v8h*)pp, *(const v8h*)(pp + 16));
#pragma unroll
      for (int j = 0; j < 4; ++j) {
        const _Float16* vp = &sVt[st][(j * 16 + cn) * 72] + c * 32 + hi * 8;
        v16h vf = combine(*(const v8h*)vp, *(const v8h*)(vp + 16));
        acc[j] = wmma_f16(pf, vf, acc[j]);
      }
    }
  }

  // ---- normalize + store f16 [b*s, d] ----
#pragma unroll
  for (int j = 0; j < 4; ++j)
#pragma unroll
    for (int r = 0; r < 8; ++r) {
      float l = lrun[r];
      float o = (l > 0.0f) ? acc[j][r] / l : 0.0f;
      size_t row = rowbase + r + hi * 8;
      O[row * D_MODEL + h * HEAD_DIM + j * 16 + cn] = (_Float16)o;
    }
}

// ---------------- launcher ----------------
extern "C" void kernel_launch(void* const* d_in, const int* in_sizes, int n_in,
                              void* d_out, int out_size, void* d_ws, size_t ws_size,
                              hipStream_t stream) {
  (void)in_sizes; (void)n_in; (void)out_size; (void)ws_size;
  const float*         x   = (const float*)d_in[0];
  const unsigned char* pad = (const unsigned char*)d_in[1];
  const float*         Wq  = (const float*)d_in[2];
  const float*         Wk  = (const float*)d_in[3];
  const float*         Wv  = (const float*)d_in[4];
  const float*         Wo  = (const float*)d_in[5];
  float*               out = (float*)d_out;

  const size_t MS = (size_t)BATCH * SEQ;        // 8192 rows
  const size_t XE = MS * D_MODEL;
  const size_t WE = (size_t)D_MODEL * D_MODEL;

  char* ws = (char*)d_ws;
  _Float16* xh  = (_Float16*)ws; ws += XE * 2;  // x, f16 [M][K]
  _Float16* wqh = (_Float16*)ws; ws += WE * 2;  // Wq^T, f16 [N][K]
  _Float16* wkh = (_Float16*)ws; ws += WE * 2;
  _Float16* wvh = (_Float16*)ws; ws += WE * 2;
  _Float16* woh = (_Float16*)ws; ws += WE * 2;
  _Float16* qh  = (_Float16*)ws; ws += XE * 2;
  _Float16* kh  = (_Float16*)ws; ws += XE * 2;
  _Float16* vh  = (_Float16*)ws; ws += XE * 2;
  _Float16* ah  = (_Float16*)ws; ws += XE * 2;

  cast_f16_kernel<<<(int)((XE + 255) / 256), 256, 0, stream>>>(x, xh, (int)XE);
  dim3 tb(32, 8), tg(D_MODEL / 32, D_MODEL / 32);
  cast_transpose_kernel<<<tg, tb, 0, stream>>>(Wq, wqh, D_MODEL, D_MODEL);
  cast_transpose_kernel<<<tg, tb, 0, stream>>>(Wk, wkh, D_MODEL, D_MODEL);
  cast_transpose_kernel<<<tg, tb, 0, stream>>>(Wv, wvh, D_MODEL, D_MODEL);
  cast_transpose_kernel<<<tg, tb, 0, stream>>>(Wo, woh, D_MODEL, D_MODEL);

  dim3 gg(D_MODEL / 128, (unsigned)(MS / 128));
  gemm_wmma_kernel<false><<<gg, 256, 0, stream>>>(xh, wqh, nullptr, qh, (int)MS, D_MODEL, D_MODEL);
  gemm_wmma_kernel<false><<<gg, 256, 0, stream>>>(xh, wkh, nullptr, kh, (int)MS, D_MODEL, D_MODEL);
  gemm_wmma_kernel<false><<<gg, 256, 0, stream>>>(xh, wvh, nullptr, vh, (int)MS, D_MODEL, D_MODEL);

  flash_attn_kernel<<<dim3(SEQ / 128, BATCH * N_HEADS), 256, 0, stream>>>(qh, kh, vh, pad, ah);

  gemm_wmma_kernel<true><<<gg, 256, 0, stream>>>(ah, woh, out, nullptr, (int)MS, D_MODEL, D_MODEL);
}
